// MLPAttention_81982335746598
// MI455X (gfx1250) — matrix operand order repaired and checked by reading, not compiled, not run on hardware
//
#include <hip/hip_runtime.h>
#include <hip/hip_bf16.h>

#define D  128
#define NQ 128
#define NK 4096
#define NE (NQ * NK)   // 524288 columns of dist

typedef __bf16 bf16x16 __attribute__((ext_vector_type(16)));
typedef float  f32x8   __attribute__((ext_vector_type(8)));
typedef float  f32x4   __attribute__((ext_vector_type(4)));

// exact pointee type of the async-to-LDS builtin: "V4i" = gcc-style int vector
typedef int v4i_raw __attribute__((vector_size(16)));
typedef __attribute__((address_space(1))) v4i_raw* g_v4i_ptr;   // global
typedef __attribute__((address_space(3))) v4i_raw* l_v4i_ptr;   // LDS

union FragU { uint4 q[2]; bf16x16 v; };   // 32 bytes = one wmma operand per lane

static __device__ __forceinline__ unsigned short f32_to_bf16(float f) {
  unsigned int u = __float_as_uint(f);
  u += 0x7FFFu + ((u >> 16) & 1u);        // round-to-nearest-even
  return (unsigned short)(u >> 16);
}

// sign-aware float atomic-min built on integer atomics (init to +FLT_MAX)
static __device__ __forceinline__ void atomicMinFloat(float* addr, float val) {
  if (val >= 0.0f) atomicMin((int*)addr, __float_as_int(val));
  else             atomicMax((unsigned int*)addr, __float_as_uint(val));
}

// ---------------------------------------------------------------- prep kernels
__global__ void init_gmin_kernel(float* gmin) { *gmin = 3.402823466e+38f; }

// qpb[c,n] = b1[c] + sum_d W1[c,d] * query[d,n]     (W1q = W1[:, :128])
__global__ void qpb_kernel(const float* __restrict__ W1, const float* __restrict__ b1,
                           const float* __restrict__ query, float* __restrict__ qpb) {
  int c = blockIdx.x, n = threadIdx.x;
  float acc = b1[c];
  for (int d = 0; d < D; ++d)
    acc = fmaf(W1[c * 384 + d], query[d * NQ + n], acc);
  qpb[c * NQ + n] = acc;
}

// kp[c,m] = sum_d W1[c,128+d] * key[d,m]            (W1k = W1[:, 128:256])
__global__ void kp_kernel(const float* __restrict__ W1, const float* __restrict__ key,
                          float* __restrict__ kp) {
  int c = blockIdx.x >> 4;
  int m = ((blockIdx.x & 15) << 8) + threadIdx.x;
  float acc = 0.0f;
  for (int d = 0; d < D; ++d)
    acc = fmaf(W1[c * 384 + 128 + d], key[d * NK + m], acc);
  kp[c * NK + m] = acc;
}

// Convert W1d (= W1[:, 256:384]) into bf16 A-fragments, pre-swizzled per the
// 16-bit A 16x32 VGPR layout (ISA 7.12.2):
//   M = lane&15 ; K = (j>>3)*16 + (lane>=16)*8 + ((j>>1)&3)*2 + (j&1)
// Abuf index = ((ct*4 + kt)*32 + lane)*16 + j  (16384 ushorts total)
__global__ void aconv_kernel(const float* __restrict__ W1, unsigned short* __restrict__ Abuf) {
  int id   = blockIdx.x * 256 + threadIdx.x;   // grid 64 x 256
  int j    = id & 15;
  int lane = (id >> 4) & 31;
  int kt   = (id >> 9) & 3;
  int ct   = id >> 11;
  int M = lane & 15;
  int K = ((j >> 3) << 4) + ((lane >> 4) << 3) + (((j >> 1) & 3) << 1) + (j & 1);
  int c = ct * 16 + M;
  int d = kt * 32 + K;
  Abuf[id] = f32_to_bf16(W1[c * 384 + 256 + d]);
}

// ---------------------------------------------------------------- main kernel
// One WG = 128(c) x 128(e) tile of dp, fused with relu/W2 epilogue -> scores.
// 8 waves; wave w owns e-strip [w*16, w*16+16). 4 k-steps x 8 c-tiles of
// v_wmma_f32_16x16x32_bf16, B fragment reused across the 8 c-tiles.
__global__ void __launch_bounds__(256)
mlp_scores_kernel(const float* __restrict__ dist,
                  const float* __restrict__ qpb,
                  const float* __restrict__ kp,
                  const unsigned short* __restrict__ Abuf,
                  const float* __restrict__ W2,
                  const float* __restrict__ b2,
                  float* __restrict__ scores,
                  float* __restrict__ gmin) {
  __shared__ unsigned short ldsA[16384];   // 32 KB: W1d fragments (all ct,kt)
  __shared__ unsigned short ldsB[16384];   // 32 KB: dist tile fragments (all strip,kt)

  const int e0   = blockIdx.x << 7;        // first of 128 columns
  const int n    = blockIdx.x >> 5;        // e0 / 4096 (row of the score matrix)
  const int m0   = (blockIdx.x & 31) << 7; // e0 % 4096
  const int t    = threadIdx.x;
  const int lane = t & 31;
  const int w    = t >> 5;

  // warm L2/WGP$ for the epilogue tables while we stage tiles
  __builtin_prefetch(&qpb[n], 0, 1);
  __builtin_prefetch(&kp[m0 + t], 0, 1);

  // ---- stage A fragments (pre-swizzled bf16 W1d, 32 KB) into LDS ----
  // Pure copy: ideal for the CDNA5 async global->LDS path (ASYNCcnt-tracked,
  // no VGPR round-trip). Fallback: register copy.
#if __has_builtin(__builtin_amdgcn_global_load_async_to_lds_b128)
  {
    const uint4* src = (const uint4*)Abuf;
    uint4*       dst = (uint4*)ldsA;
    #pragma unroll
    for (int i = 0; i < 8; ++i) {
      __builtin_amdgcn_global_load_async_to_lds_b128(
          (g_v4i_ptr)(src + t + 256 * i),
          (l_v4i_ptr)(dst + t + 256 * i),
          0, 0);
    }
  }
#else
  {
    const uint4* src = (const uint4*)Abuf;
    uint4*       dst = (uint4*)ldsA;
    #pragma unroll
    for (int i = 0; i < 8; ++i) dst[t + 256 * i] = src[t + 256 * i];
  }
#endif

  // ---- stage B: 128(d) x 128(e) fp32 dist tile -> bf16, swizzled ----
  // dist is streamed exactly once (268 MB total): use non-temporal loads so it
  // does not evict the reused tables (kp/qpb/value/scores) from the 192MB L2.
  #pragma unroll
  for (int i = 0; i < 16; ++i) {
    int d  = (t >> 5) + (i << 3);
    int el = (t & 31) << 2;
    f32x4 v = __builtin_nontemporal_load(
        (const f32x4*)(dist + (size_t)d * NE + e0 + el));
    int kt = d >> 5, j = (d & 7) | (((d >> 4) & 1) << 3), khalf = (d >> 3) & 1;
    int base = ((((el >> 4) << 2) + kt) * 32 + (el & 15) + (khalf << 4)) * 16 + j;
    ldsB[base     ] = f32_to_bf16(v.x);
    ldsB[base + 16] = f32_to_bf16(v.y);
    ldsB[base + 32] = f32_to_bf16(v.z);
    ldsB[base + 48] = f32_to_bf16(v.w);
  }

#if __has_builtin(__builtin_amdgcn_global_load_async_to_lds_b128)
  // async A-copy must land in LDS before the workgroup barrier
#  if __has_builtin(__builtin_amdgcn_s_wait_asynccnt)
  __builtin_amdgcn_s_wait_asynccnt(0);
#  else
  asm volatile("s_wait_asynccnt 0x0" ::: "memory");
#  endif
#endif
  __syncthreads();

  f32x8 acc[8];
  const f32x8 zero = {0.f, 0.f, 0.f, 0.f, 0.f, 0.f, 0.f, 0.f};
  #pragma unroll
  for (int ct = 0; ct < 8; ++ct) acc[ct] = zero;

  #pragma unroll
  for (int kt = 0; kt < 4; ++kt) {
    FragU b;
    const uint4* pb = (const uint4*)&ldsB[((((w << 2) + kt) * 32 + lane) << 4)];
    b.q[0] = pb[0]; b.q[1] = pb[1];
    #pragma unroll
    for (int ct = 0; ct < 8; ++ct) {
      FragU a;
      const uint4* pa = (const uint4*)&ldsA[((((ct << 2) + kt) * 32 + lane) << 4)];
      a.q[0] = pa[0]; a.q[1] = pa[1];
      acc[ct] = __builtin_amdgcn_wmma_f32_16x16x32_bf16(
          false, a.v, false, b.v, (short)0, acc[ct], false, false);
    }
  }

  // epilogue: C/D layout -> lane owns column N = lane&15 of its strip,
  // rows M = v + 8*(lane>=16). Pair of lanes (l, l^16) covers all 128 c.
  const int   hi8  = (lane >> 4) << 3;
  const int   mcol = m0 + (w << 4) + (lane & 15);
  const float b2v  = b2[0];
  float partial = 0.0f;
  #pragma unroll
  for (int ct = 0; ct < 8; ++ct) {
    #pragma unroll
    for (int v = 0; v < 8; ++v) {
      int   c = (ct << 4) + v + hi8;
      float x = acc[ct][v] + qpb[c * NQ + n] + kp[c * NK + mcol];
      partial = fmaf(W2[c], fmaxf(x, 0.0f), partial);
    }
  }
  float score = partial + __shfl_xor(partial, 16, 32) + b2v;
  if ((lane & 16) == 0) scores[n * NK + mcol] = score;   // pre-mask scores

  float s = score;
  #pragma unroll
  for (int off = 8; off; off >>= 1) s = fminf(s, __shfl_xor(s, off, 32));
  if (lane == 0) atomicMinFloat(gmin, s);
}

// ------------------------------------------------- mask + softmax (per row n)
__global__ void __launch_bounds__(256)
mask_softmax_kernel(float* __restrict__ scores, const unsigned char* __restrict__ mask,
                    const float* __restrict__ gmin, float* __restrict__ prob) {
  __shared__ float red[256];
  const int n = blockIdx.x, t = threadIdx.x;
  const float gm = *gmin - 20.0f;

  float vals[16];
  float lmax = -3.402823466e+38f;
  #pragma unroll
  for (int i = 0; i < 16; ++i) {
    int m = i * 256 + t;
    float s = scores[n * NK + m];
    unsigned char mk = mask[n * NK + m];
    float sm = s + (mk ? 0.0f : gm);
    vals[i] = sm;
    scores[n * NK + m] = sm;                 // returned scores are masked
    lmax = fmaxf(lmax, sm);
  }
  red[t] = lmax; __syncthreads();
  for (int o = 128; o > 0; o >>= 1) { if (t < o) red[t] = fmaxf(red[t], red[t + o]); __syncthreads(); }
  float rmax = red[0]; __syncthreads();

  float lsum = 0.0f;
  #pragma unroll
  for (int i = 0; i < 16; ++i) { vals[i] = __expf(vals[i] - rmax); lsum += vals[i]; }
  red[t] = lsum; __syncthreads();
  for (int o = 128; o > 0; o >>= 1) { if (t < o) red[t] += red[t + o]; __syncthreads(); }
  float inv = 1.0f / red[0];
  #pragma unroll
  for (int i = 0; i < 16; ++i) prob[n * NK + i * 256 + t] = vals[i] * inv;
}

// ---------------------------------------------- out[d,n] = sum_m prob[n,m]*value[d,m]
__global__ void out_gemm_kernel(const float* __restrict__ prob,
                                const float* __restrict__ value,
                                float* __restrict__ out) {
  const int n = blockIdx.x, d = threadIdx.x;   // 128 x 128
  const float4* p4 = (const float4*)(prob + (size_t)n * NK);
  const float4* v4 = (const float4*)(value + (size_t)d * NK);
  float acc = 0.0f;
  for (int i = 0; i < NK / 4; ++i) {
    float4 p = p4[i], v = v4[i];
    acc += p.x * v.x + p.y * v.y + p.z * v.z + p.w * v.w;
  }
  out[d * NQ + n] = acc;
}

// ---------------------------------------------------------------- launcher
extern "C" void kernel_launch(void* const* d_in, const int* in_sizes, int n_in,
                              void* d_out, int out_size, void* d_ws, size_t ws_size,
                              hipStream_t stream) {
  const float* query = (const float*)d_in[0];
  const float* key_  = (const float*)d_in[1];
  const float* value = (const float*)d_in[2];
  const float* dist  = (const float*)d_in[3];
  const unsigned char* mask = (const unsigned char*)d_in[4];  // numpy bool = 1 byte
  const float* W1 = (const float*)d_in[5];
  const float* b1 = (const float*)d_in[6];
  const float* W2 = (const float*)d_in[7];
  const float* b2 = (const float*)d_in[8];

  float* out    = (float*)d_out;        // [0, 16384): out (d,n)
  float* scores = out + D * NQ;         // [16384, 540672): scores (n,m)

  float* ws   = (float*)d_ws;
  float* qpb  = ws;                                           // 16384 f
  float* kp   = qpb + D * NQ;                                 // 524288 f
  unsigned short* Abuf = (unsigned short*)(kp + (size_t)D * NK);  // 16384 us
  float* prob = (float*)(Abuf + 16384);                       // 524288 f
  float* gmin = prob + (size_t)NQ * NK;                       // 1 f

  init_gmin_kernel<<<1, 1, 0, stream>>>(gmin);
  qpb_kernel<<<D, NQ, 0, stream>>>(W1, b1, query, qpb);
  kp_kernel<<<D * (NK / 256), 256, 0, stream>>>(W1, key_, kp);
  aconv_kernel<<<64, 256, 0, stream>>>(W1, Abuf);
  mlp_scores_kernel<<<NE / 128, 256, 0, stream>>>(dist, qpb, kp, Abuf, W2, b2, scores, gmin);
  mask_softmax_kernel<<<NQ, 256, 0, stream>>>(scores, mask, gmin, prob);
  out_gemm_kernel<<<NQ, D, 0, stream>>>(prob, value, out);
}
